// GeometricReasoningOriginalImpl_81277961109590
// MI455X (gfx1250) — compile-verified, matrix operand order checked by reading, not run on hardware
//
#include <hip/hip_runtime.h>
#include <hip/hip_bf16.h>
#include <cstdint>

// ---------------------------------------------------------------------------
// GeometricReasoning (IPA-style attention) for MI455X / gfx1250, wave32.
//
// Pipeline (all on `stream`):
//   0) prep_weights : w_proj -> f16 ; w_out -> f16 [c][64] (K padded 48->64)
//   1) ln_proj      : LayerNorm rows of s -> f16 in LDS; w_proj K-chunks
//                     staged into LDS by the Tensor Data Mover (TDM,
//                     double-buffered, TENSORcnt-tracked), then WMMA GEMM
//                     (M=2048,N=240,K=1024) p = ns @ w_proj^T      [v_wmma]
//   2) geom         : per (b,s,h) apply 3x3 rot (+trans) -> Qr,Kr,V,Qd,Kd
//   3) attn         : one query per wave32, online softmax over k=1024
//   4) rotback      : out = rot^T @ attn_out, mask, -> f16 [2048][64]
//   5) out_gemm     : WMMA GEMM (M=2048,N=1024,K=64) -> d_out       [v_wmma]
// ---------------------------------------------------------------------------

typedef __attribute__((ext_vector_type(16))) _Float16 v16h;
typedef __attribute__((ext_vector_type(8)))  float    v8f;
typedef __attribute__((ext_vector_type(4)))  unsigned int u32x4;
typedef __attribute__((ext_vector_type(4)))  unsigned int v4u;
typedef __attribute__((ext_vector_type(8)))  int v8i;
typedef __attribute__((ext_vector_type(4)))  int v4i;

#define SQ3INV 0.57735026918962576f
#define NEGF  -3.4028234663852886e38f

constexpr int Bb = 2, S = 1024, Cs = 1024, H = 16;
constexpr int NROWS = Bb * S;      // 2048
constexpr int DPROJ = 240;
constexpr int KOUT  = 64;          // 48 padded to 64 for K%32==0
constexpr int NKT   = Cs / 32;     // 32 K-chunks in the projection GEMM

// workspace layout (bytes); total ~10.2 MB
constexpr size_t OFF_NS    = 0;                                   // reserved
constexpr size_t OFF_WPROJ = OFF_NS    + (size_t)NROWS*Cs*2;      // f16 [240][1024]
constexpr size_t OFF_P     = OFF_WPROJ + (size_t)DPROJ*Cs*2;      // f32 [2048][240]
constexpr size_t SZ_BH4    = (size_t)Bb*H*S*16;                   // float4 [B*H][S]
constexpr size_t OFF_QR    = OFF_P  + (size_t)NROWS*DPROJ*4;
constexpr size_t OFF_KR    = OFF_QR + SZ_BH4;
constexpr size_t OFF_VV    = OFF_KR + SZ_BH4;
constexpr size_t OFF_QD    = OFF_VV + SZ_BH4;
constexpr size_t OFF_KD    = OFF_QD + SZ_BH4;
constexpr size_t OFF_AO    = OFF_KD + SZ_BH4;                     // float4 [2048][16]
constexpr size_t OFF_O48   = OFF_AO + (size_t)NROWS*H*16;         // f16 [2048][64]
constexpr size_t OFF_WOUT  = OFF_O48 + (size_t)NROWS*KOUT*2;      // f16 [1024][64]

#if __has_builtin(__builtin_amdgcn_tensor_load_to_lds) && \
    __has_builtin(__builtin_amdgcn_s_wait_tensorcnt)
#define HAVE_TDM 1
#else
#define HAVE_TDM 0
#endif

__device__ __forceinline__ float wave_sum(float v) {
  for (int o = 16; o; o >>= 1) v += __shfl_xor(v, o, 32);
  return v;
}
__device__ __forceinline__ float wave_max(float v) {
  for (int o = 16; o; o >>= 1) v = fmaxf(v, __shfl_xor(v, o, 32));
  return v;
}

#if HAVE_TDM
// Issue one TDM transfer: a 32(k) x 240(d) f16 tile of w_proj (viewed as a
// 1024 x 240 tensor, dim0 = k contiguous) into LDS at lds_addr, packed as
// 240 rows of 32 halfs. D# per CDNA5 ISA ch.8 (2-D: groups 2/3 zero).
__device__ __forceinline__ void tdm_load_wproj_chunk(unsigned int lds_addr,
                                                     const _Float16* gtile) {
  unsigned long long ga = (unsigned long long)(uintptr_t)gtile;
  v4u g0;
  g0[0] = 1u;                                  // count=1, user mode
  g0[1] = lds_addr;                            // lds_addr (bytes)
  g0[2] = (unsigned int)ga;                    // global_addr[31:0]
  g0[3] = (unsigned int)(ga >> 32) | (2u << 30); // global_addr[56:32] | type=2
  v8i g1;
  g1[0] = (int)(1u << 16);                     // wg_mask=0, data_size=1 (2B)
  g1[1] = (int)(1024u << 16);                  // tensor_dim0 = 1024 (low16)
  g1[2] = (int)(240u << 16);                   // dim0 hi=0 | tensor_dim1=240
  g1[3] = (int)(32u << 16);                    // dim1 hi=0 | tile_dim0=32
  g1[4] = (int)240u;                           // tile_dim1=240 | tile_dim2=0
  g1[5] = (int)1024u;                          // tensor_dim0_stride = 1024
  g1[6] = 0;                                   // stride0 hi | stride1 lo
  g1[7] = 0;                                   // stride1 hi
  v4i z4 = {0, 0, 0, 0};
#if defined(__clang_major__) && __clang_major__ >= 23
  v8i z8 = {0, 0, 0, 0, 0, 0, 0, 0};
  __builtin_amdgcn_tensor_load_to_lds(g0, g1, z4, z4, z8, 0);
#else
  __builtin_amdgcn_tensor_load_to_lds(g0, g1, z4, z4, 0);
#endif
}
#endif

// --------------------------- kernel 0: weight prep -------------------------
__global__ void prep_weights(const float* __restrict__ wproj,
                             const float* __restrict__ wout,
                             _Float16* __restrict__ wproj_h,
                             _Float16* __restrict__ wout_h) {
  int i = blockIdx.x * blockDim.x + threadIdx.x;
  const int nproj = DPROJ * Cs;
  if (i < nproj) {
    wproj_h[i] = (_Float16)wproj[i];
  } else {
    int j = i - nproj;                     // wout: [c][k], k padded to 64
    if (j < Cs * KOUT) {
      int c = j >> 6, k = j & 63;
      wout_h[c * KOUT + k] = (k < 48) ? (_Float16)wout[c * 48 + k] : (_Float16)0.f;
    }
  }
}

// ------------------- kernel 1: LayerNorm + projection GEMM -----------------
// 128 blocks of 256 threads (8 waves). Each block: 16 rows of ns (f16 in LDS)
// x 240 cols of p. Wave w owns N-tiles {w, w+8}. w_proj K-chunks (240x32 f16,
// 15 KB) are DMA'd into a double-buffered LDS tile by the TDM while the
// previous chunk feeds v_wmma_f32_16x16x32_f16.
// Fragment layout per ISA 7.12.2 (16-bit A 16x32): lane m=lane&15,
// kb=(lane<16?0:8), halves [k0..k0+7] and [k0+16..k0+23] -> two b128 loads.
__global__ __launch_bounds__(256) void ln_proj(const float* __restrict__ s,
                                               const float* __restrict__ ln_w,
                                               const _Float16* __restrict__ wproj_h,
                                               float* __restrict__ p) {
  __shared__ _Float16 lds_ns[16][Cs];                       // 32 KB
  __shared__ __align__(16) _Float16 ldsB[2][DPROJ * 32];    // 2 x 15 KB
  const int lane = threadIdx.x & 31, wave = threadIdx.x >> 5;
  const int rowbase = blockIdx.x * 16;

#if HAVE_TDM
  if (wave == 0)   // prologue: start DMA of chunk 0 while everyone does LN
    tdm_load_wproj_chunk((unsigned int)(uintptr_t)&ldsB[0][0], wproj_h);
#endif

  // LayerNorm: each wave does 2 rows
  for (int rr = 0; rr < 2; ++rr) {
    int r = wave * 2 + rr;
    const float* rp = s + (size_t)(rowbase + r) * Cs;
    float sum = 0.f, sq = 0.f;
    for (int i = lane; i < Cs; i += 32) {
      __builtin_prefetch(rp + i + 256, 0, 1);
      float x = rp[i]; sum += x; sq += x * x;
    }
    sum = wave_sum(sum); sq = wave_sum(sq);
    float mu = sum * (1.f / Cs);
    float var = sq * (1.f / Cs) - mu * mu;
    float rstd = rsqrtf(var + 1e-5f);
    for (int i = lane; i < Cs; i += 32)
      lds_ns[r][i] = (_Float16)((rp[i] - mu) * rstd * ln_w[i]);
  }

  const int m   = lane & 15;
  const int kb  = (lane < 16) ? 0 : 8;
  const int nt0 = wave, nt1 = wave + 8;       // nt1==15 is skipped
  v8f acc0 = {}, acc1 = {};

  for (int kt = 0; kt < NKT; ++kt) {
    int buf = kt & 1;
#if HAVE_TDM
    if (wave == 0) {
      if (kt + 1 < NKT) {
        tdm_load_wproj_chunk((unsigned int)(uintptr_t)&ldsB[buf ^ 1][0],
                             wproj_h + (size_t)(kt + 1) * 32);
        __builtin_amdgcn_s_wait_tensorcnt(1);  // in-order: chunk kt done
      } else {
        __builtin_amdgcn_s_wait_tensorcnt(0);
      }
    }
#else
    // fallback: cooperative staging of chunk kt (960 x 16B)
    for (int i = threadIdx.x; i < DPROJ * 4; i += 256) {
      int d = i >> 2, seg = (i & 3) * 8;
      *reinterpret_cast<u32x4*>(&ldsB[buf][d * 32 + seg]) =
          *reinterpret_cast<const u32x4*>(wproj_h + (size_t)d * Cs + kt * 32 + seg);
    }
#endif
    __syncthreads();   // chunk kt visible (also covers LN on first iter)

    int k0 = kt * 32 + kb;
    union { v16h h; u32x4 u[2]; } fa, fb;
    fa.u[0] = *reinterpret_cast<const u32x4*>(&lds_ns[m][k0]);
    fa.u[1] = *reinterpret_cast<const u32x4*>(&lds_ns[m][k0 + 16]);

    fb.u[0] = *reinterpret_cast<const u32x4*>(&ldsB[buf][(nt0 * 16 + m) * 32 + kb]);
    fb.u[1] = *reinterpret_cast<const u32x4*>(&ldsB[buf][(nt0 * 16 + m) * 32 + kb + 16]);
    acc0 = __builtin_amdgcn_wmma_f32_16x16x32_f16(false, fa.h, false, fb.h,
                                                  (short)0, acc0, false, false);
    if (nt1 < 15) {   // wave-uniform: EXEC stays all-ones inside
      fb.u[0] = *reinterpret_cast<const u32x4*>(&ldsB[buf][(nt1 * 16 + m) * 32 + kb]);
      fb.u[1] = *reinterpret_cast<const u32x4*>(&ldsB[buf][(nt1 * 16 + m) * 32 + kb + 16]);
      acc1 = __builtin_amdgcn_wmma_f32_16x16x32_f16(false, fa.h, false, fb.h,
                                                    (short)0, acc1, false, false);
    }
    __syncthreads();   // protect ldsB[buf] before it is re-filled at kt+1
  }

  // D layout: VGPR r -> M = r + (lane<16?0:8); N = lane&15
  int rout = rowbase + ((lane < 16) ? 0 : 8);
  for (int r = 0; r < 8; ++r)
    p[(size_t)(rout + r) * DPROJ + nt0 * 16 + m] = acc0[r];
  if (nt1 < 15)
    for (int r = 0; r < 8; ++r)
      p[(size_t)(rout + r) * DPROJ + nt1 * 16 + m] = acc1[r];
}

// ----------------- kernel 2: per-residue rotations -------------------------
__global__ void geom(const float* __restrict__ p, const float* __restrict__ rot,
                     const float* __restrict__ trans,
                     float4* __restrict__ QR, float4* __restrict__ KR,
                     float4* __restrict__ VV, float4* __restrict__ QD,
                     float4* __restrict__ KD) {
  int idx = blockIdx.x * blockDim.x + threadIdx.x;   // (b*S)*H threads
  if (idx >= NROWS * H) return;
  int h = idx & 15, bs = idx >> 4;
  int b = bs >> 10, si = bs & 1023;
  const float* R  = rot + (size_t)bs * 9;
  const float* pr = p + (size_t)bs * DPROJ;
  float r0 = R[0], r1 = R[1], r2 = R[2], r3 = R[3], r4 = R[4],
        r5 = R[5], r6 = R[6], r7 = R[7], r8 = R[8];
  size_t o = (size_t)((b * H + h) * S + si);
#define ROT3(px, dst)                                                     \
  { float x = (px)[0], y = (px)[1], z = (px)[2];                          \
    dst = make_float4(r0*x + r1*y + r2*z, r3*x + r4*y + r5*z,             \
                      r6*x + r7*y + r8*z, 0.f); }
  float4 v;
  ROT3(pr + h * 3,        v); QR[o] = v;                 // q_rot  (j = h)
  ROT3(pr + 48 + h * 3,   v); KR[o] = v;                 // k_rot  (j = 16+h)
  ROT3(pr + 96 + h * 3,   v); VV[o] = v;                 // value  (j = 32+h)
  float tx = trans[(size_t)bs * 3 + 0], ty = trans[(size_t)bs * 3 + 1],
        tz = trans[(size_t)bs * 3 + 2];
  ROT3(pr + 144 + h * 3,  v);
  QD[o] = make_float4(v.x + tx, v.y + ty, v.z + tz, 0.f); // q_dist
  ROT3(pr + 192 + h * 3,  v);
  KD[o] = make_float4(v.x + tx, v.y + ty, v.z + tz, 0.f); // k_dist
#undef ROT3
}

// ----------------- kernel 3: geometric attention (flash-style) -------------
// One query per wave32; online softmax over k=1024 in 32-wide chunks.
__global__ __launch_bounds__(256) void attn(const float4* __restrict__ QR,
                                            const float4* __restrict__ KR,
                                            const float4* __restrict__ VV,
                                            const float4* __restrict__ QD,
                                            const float4* __restrict__ KD,
                                            const float* __restrict__ dist_scale,
                                            const float* __restrict__ rot_scale,
                                            const unsigned char* __restrict__ amask,
                                            const int* __restrict__ seqid,
                                            const int* __restrict__ chainid,
                                            float4* __restrict__ AO) {
  const int lane = threadIdx.x & 31, wave = threadIdx.x >> 5;
  const int bh = blockIdx.x >> 7;                    // 0..31
  const int q  = ((blockIdx.x & 127) << 3) + wave;   // 0..1023
  const int b = bh >> 4, h = bh & 15;
  float dsv = dist_scale[h], rsv = rot_scale[h];
  float dw = (dsv > 20.f) ? dsv : log1pf(__expf(dsv));   // softplus
  float rw = (rsv > 20.f) ? rsv : log1pf(__expf(rsv));
  const size_t base = (size_t)bh * S;
  float4 qr = QR[base + q], qd = QD[base + q];
  int sq = seqid[b * S + q], cq = chainid[b * S + q];

  float mrun = NEGF, lrun = 0.f, a0 = 0.f, a1 = 0.f, a2 = 0.f;
  for (int kc = 0; kc < S / 32; ++kc) {
    int k = kc * 32 + lane;
    float4 kr = KR[base + k], kd = KD[base + k], vv = VV[base + k];
    float rt = (qr.x * kr.x + qr.y * kr.y + qr.z * kr.z) * SQ3INV;
    float dx = qd.x - kd.x, dy = qd.y - kd.y, dz = qd.z - kd.z;
    float dt = sqrtf(dx * dx + dy * dy + dz * dz) * SQ3INV;
    float bias;
    if (chainid[b * S + k] != cq || !amask[b * S + k]) bias = NEGF;
    else bias = (seqid[b * S + k] == sq) ? 1.f : 0.f;
    float sc = rt * rw - dt * dw + bias;
    float mn = fmaxf(mrun, wave_max(sc));
    float corr = __expf(mrun - mn);
    float pe = __expf(sc - mn);
    lrun = lrun * corr + pe;
    a0 = a0 * corr + pe * vv.x;
    a1 = a1 * corr + pe * vv.y;
    a2 = a2 * corr + pe * vv.z;
    mrun = mn;
  }
  lrun = wave_sum(lrun);
  a0 = wave_sum(a0); a1 = wave_sum(a1); a2 = wave_sum(a2);
  if (lane == 0) {
    float inv = 1.f / lrun;
    AO[(size_t)(b * S + q) * H + h] = make_float4(a0 * inv, a1 * inv, a2 * inv, 0.f);
  }
}

// ----------------- kernel 4: rotate back, mask, pack f16 -------------------
__global__ void rotback(const float4* __restrict__ AO, const float* __restrict__ rot,
                        const unsigned char* __restrict__ amask,
                        _Float16* __restrict__ o48) {
  int idx = blockIdx.x * blockDim.x + threadIdx.x;
  if (idx >= NROWS * H) return;
  int h = idx & 15, bs = idx >> 4;
  const float* R = rot + (size_t)bs * 9;
  float4 a = AO[(size_t)bs * H + h];
  // out_i = sum_j rot[j][i] * in_j   ('bsji')
  float o0 = R[0] * a.x + R[3] * a.y + R[6] * a.z;
  float o1 = R[1] * a.x + R[4] * a.y + R[7] * a.z;
  float o2 = R[2] * a.x + R[5] * a.y + R[8] * a.z;
  float msk = amask[bs] ? 1.f : 0.f;
  _Float16* row = o48 + (size_t)bs * KOUT;
  row[h * 3 + 0] = (_Float16)(o0 * msk);
  row[h * 3 + 1] = (_Float16)(o1 * msk);
  row[h * 3 + 2] = (_Float16)(o2 * msk);
  if (h == 0)
    for (int k = 48; k < KOUT; ++k) row[k] = (_Float16)0.f;
}

// ----------------- kernel 5: output GEMM (WMMA) ----------------------------
// M=2048, N=1024, K=64. Block = 8 waves; each wave owns one 16-col tile.
__global__ __launch_bounds__(256) void out_gemm(const _Float16* __restrict__ o48,
                                                const _Float16* __restrict__ wout_h,
                                                float* __restrict__ out) {
  const int lane = threadIdx.x & 31, wave = threadIdx.x >> 5;
  const int rowbase = blockIdx.x * 16;
  const int nt = blockIdx.y * 8 + wave;    // 0..63
  const int m = lane & 15;
  const int kb = (lane < 16) ? 0 : 8;
  const _Float16* abase = o48 + (size_t)(rowbase + m) * KOUT;
  const _Float16* bbase = wout_h + (size_t)(nt * 16 + m) * KOUT;
  v8f acc = {};
  for (int kt = 0; kt < KOUT / 32; ++kt) {
    int k0 = kt * 32 + kb;
    union { v16h h; u32x4 u[2]; } fa, fb;
    fa.u[0] = *reinterpret_cast<const u32x4*>(abase + k0);
    fa.u[1] = *reinterpret_cast<const u32x4*>(abase + k0 + 16);
    fb.u[0] = *reinterpret_cast<const u32x4*>(bbase + k0);
    fb.u[1] = *reinterpret_cast<const u32x4*>(bbase + k0 + 16);
    acc = __builtin_amdgcn_wmma_f32_16x16x32_f16(false, fa.h, false, fb.h,
                                                 (short)0, acc, false, false);
  }
  int rout = rowbase + ((lane < 16) ? 0 : 8);
  int col  = nt * 16 + m;
  for (int r = 0; r < 8; ++r)
    out[(size_t)(rout + r) * Cs + col] = acc[r];
}

// ---------------------------------------------------------------------------
extern "C" void kernel_launch(void* const* d_in, const int* in_sizes, int n_in,
                              void* d_out, int out_size, void* d_ws, size_t ws_size,
                              hipStream_t stream) {
  const float* s          = (const float*)d_in[0];
  const float* rot        = (const float*)d_in[1];
  const float* trans      = (const float*)d_in[2];
  const float* ln_w       = (const float*)d_in[3];
  const float* w_proj     = (const float*)d_in[4];
  const float* w_out      = (const float*)d_in[5];
  const float* dist_scale = (const float*)d_in[6];
  const float* rot_scale  = (const float*)d_in[7];
  const unsigned char* am = (const unsigned char*)d_in[8];   // jax bool = 1 byte
  const int* seqid        = (const int*)d_in[9];
  const int* chainid      = (const int*)d_in[10];

  char* ws = (char*)d_ws;
  _Float16* wproj_h = (_Float16*)(ws + OFF_WPROJ);
  float*    p       = (float*)   (ws + OFF_P);
  float4*   QR      = (float4*)  (ws + OFF_QR);
  float4*   KR      = (float4*)  (ws + OFF_KR);
  float4*   VV      = (float4*)  (ws + OFF_VV);
  float4*   QD      = (float4*)  (ws + OFF_QD);
  float4*   KD      = (float4*)  (ws + OFF_KD);
  float4*   AO      = (float4*)  (ws + OFF_AO);
  _Float16* o48     = (_Float16*)(ws + OFF_O48);
  _Float16* wout_h  = (_Float16*)(ws + OFF_WOUT);

  const int prep_elems = DPROJ * Cs + Cs * KOUT;             // 311296
  prep_weights<<<(prep_elems + 255) / 256, 256, 0, stream>>>(w_proj, w_out, wproj_h, wout_h);
  ln_proj<<<NROWS / 16, 256, 0, stream>>>(s, ln_w, wproj_h, p);
  geom<<<(NROWS * H) / 256, 256, 0, stream>>>(p, rot, trans, QR, KR, VV, QD, KD);
  attn<<<Bb * H * (S / 8), 256, 0, stream>>>(QR, KR, VV, QD, KD,
                                             dist_scale, rot_scale, am, seqid, chainid, AO);
  rotback<<<(NROWS * H) / 256, 256, 0, stream>>>(AO, rot, am, o48);
  out_gemm<<<dim3(NROWS / 16, Cs / 128), 256, 0, stream>>>(o48, wout_h, (float*)d_out);
}
